// DrugTokenizer_56908316672426
// MI455X (gfx1250) — matrix operand order repaired and checked by reading, not compiled, not run on hardware
//
#include <hip/hip_runtime.h>

#define NNODES 50000
#define NEDGES 800000

typedef __attribute__((ext_vector_type(16))) __bf16        v16bf;
typedef __attribute__((ext_vector_type(8)))  float         v8f;
typedef __attribute__((ext_vector_type(4)))  unsigned int  u32x4;

union Frag { v16bf v; u32x4 q[2]; };

__device__ __forceinline__ unsigned short f2bf(float f) {
    unsigned int u = __float_as_uint(f);
    u += 0x7FFFu + ((u >> 16) & 1u);          // round-to-nearest-even
    return (unsigned short)(u >> 16);
}

// ---------------- setup kernels ----------------
__global__ void k_deg_init(float* deg, int n) {
    int i = blockIdx.x * blockDim.x + threadIdx.x;
    if (i < n) deg[i] = 1.0f;                 // self-loop contributes 1
}

__global__ void k_deg_accum(const int* __restrict__ dst, float* deg, int e) {
    int i = blockIdx.x * blockDim.x + threadIdx.x;
    if (i < e) unsafeAtomicAdd(&deg[dst[i]], 1.0f);
}

__global__ void k_rsqrt(float* d, int n) {
    int i = blockIdx.x * blockDim.x + threadIdx.x;
    if (i < n) d[i] = rsqrtf(d[i]);
}

__global__ void k_norm(const int* __restrict__ src, const int* __restrict__ dst,
                       const float* __restrict__ dis, float* nrm, int e) {
    int i = blockIdx.x * blockDim.x + threadIdx.x;
    if (i < e) nrm[i] = dis[src[i]] * dis[dst[i]];
}

__global__ void k_cvt(const float* __restrict__ x, unsigned short* __restrict__ xb, long n) {
    long i = (long)blockIdx.x * blockDim.x + threadIdx.x;
    if (i < n) xb[i] = f2bf(x[i]);
}

// convert K x N row-major f32 weight -> N x K bf16 (transposed, contiguous along K)
__global__ void k_wt(const float* __restrict__ W, unsigned short* __restrict__ wt, int K, int Nout) {
    int i = blockIdx.x * blockDim.x + threadIdx.x;
    if (i < K * Nout) {
        int k = i / Nout, n = i - k * Nout;
        wt[(long)n * K + k] = f2bf(W[i]);
    }
}

// ---------------- WMMA GEMM, 1x4 register-blocked ----------------
// D[M,N] = A[M,K](bf16) * B (wt pre-transposed to N-major)
// One wave computes a 16x64 strip: 1 M-tile x 4 N-tiles.
// Per K-step of 32: 2 A b128 loads + 8 B b128 loads feed 4 WMMAs.
__global__ __launch_bounds__(256) void k_gemm(const unsigned short* __restrict__ A,
                                              const unsigned short* __restrict__ Bt,
                                              float* __restrict__ D,
                                              int M, int K, int Nout) {
    int wid  = (int)((blockIdx.x * blockDim.x + threadIdx.x) >> 5);
    int lane = threadIdx.x & 31;
    int strips = Nout >> 6;                   // 64-wide strips
    int mtile  = wid / strips;
    int strip  = wid - mtile * strips;
    if (mtile * 16 >= M) return;

    int half = lane >> 4;                     // K-half selector (contiguous order: k = 16*half + s)
    int l16  = lane & 15;

    const unsigned short* arow = A  + (size_t)(mtile * 16 + l16) * K + half * 16;
    const unsigned short* brow = Bt + (size_t)(strip * 64 + l16) * K + half * 16;
    const size_t bts = (size_t)16 * K;        // stride between adjacent N-tiles in Bt

    v8f acc0 = {}, acc1 = {}, acc2 = {}, acc3 = {};
    for (int k = 0; k < K; k += 32) {
        Frag a;
        a.q[0] = *(const u32x4*)(arow + k);            // K = k+half*16   .. +7
        a.q[1] = *(const u32x4*)(arow + k + 8);        // K = k+half*16+8 .. +15

        Frag b0, b1, b2, b3;
        const unsigned short* bk = brow + k;
        b0.q[0] = *(const u32x4*)(bk);                 // K = k+half*16   .. +7
        b0.q[1] = *(const u32x4*)(bk + 8);             // K = k+half*16+8 .. +15
        b1.q[0] = *(const u32x4*)(bk + bts);
        b1.q[1] = *(const u32x4*)(bk + bts + 8);
        b2.q[0] = *(const u32x4*)(bk + 2 * bts);
        b2.q[1] = *(const u32x4*)(bk + 2 * bts + 8);
        b3.q[0] = *(const u32x4*)(bk + 3 * bts);
        b3.q[1] = *(const u32x4*)(bk + 3 * bts + 8);

        acc0 = __builtin_amdgcn_wmma_f32_16x16x32_bf16(false, a.v, false, b0.v, (short)0, acc0, false, false);
        acc1 = __builtin_amdgcn_wmma_f32_16x16x32_bf16(false, a.v, false, b1.v, (short)0, acc1, false, false);
        acc2 = __builtin_amdgcn_wmma_f32_16x16x32_bf16(false, a.v, false, b2.v, (short)0, acc2, false, false);
        acc3 = __builtin_amdgcn_wmma_f32_16x16x32_bf16(false, a.v, false, b3.v, (short)0, acc3, false, false);
    }

    // C/D layout: lanes 0-15 -> M = r, lanes 16-31 -> M = r+8; N = lane%16
    float* dbase = D + (size_t)(mtile * 16 + half * 8) * Nout + strip * 64 + l16;
#pragma unroll
    for (int r = 0; r < 8; ++r) {
        float* dr = dbase + (size_t)r * Nout;
        dr[0]  = acc0[r];
        dr[16] = acc1[r];
        dr[32] = acc2[r];
        dr[48] = acc3[r];
    }
}

// ---------------- aggregation ----------------
// init agg with the self-loop term: agg[i,:] = h[i,:] * dis[i]^2
__global__ void k_selfloop(const float* __restrict__ h, const float* __restrict__ dis,
                           float* __restrict__ agg, int n, int Nout) {
    long idx = (long)blockIdx.x * blockDim.x + threadIdx.x;
    long total = (long)n * Nout;
    if (idx < total) {
        int i = (int)(idx / Nout);
        float d = dis[i];
        agg[idx] = h[idx] * d * d;
    }
}

// one wave per edge: agg[dst,:] += h[src,:] * norm[e]
__global__ __launch_bounds__(256) void k_scatter(const int* __restrict__ src,
                                                 const int* __restrict__ dst,
                                                 const float* __restrict__ nrm,
                                                 const float* __restrict__ h,
                                                 float* __restrict__ agg,
                                                 int e, int Nout) {
    int wid  = (int)((blockIdx.x * blockDim.x + threadIdx.x) >> 5);
    int lane = threadIdx.x & 31;
    if (wid >= e) return;
    int s = src[wid], d = dst[wid];
    float w = nrm[wid];
    const float* hr = h + (size_t)s * Nout;
    float* ar = agg + (size_t)d * Nout;
    for (int j = lane; j < Nout; j += 32)
        unsafeAtomicAdd(&ar[j], hr[j] * w);
}

__global__ void k_bias(const float* __restrict__ agg, const float* __restrict__ bias,
                       float* __restrict__ out, long total, int Nout, int relu) {
    long idx = (long)blockIdx.x * blockDim.x + threadIdx.x;
    if (idx < total) {
        float v = agg[idx] + bias[idx & (Nout - 1)];   // Nout is a power of two
        out[idx] = relu ? fmaxf(v, 0.0f) : v;
    }
}

// ---------------- host ----------------
extern "C" void kernel_launch(void* const* d_in, const int* in_sizes, int n_in,
                              void* d_out, int out_size, void* d_ws, size_t ws_size,
                              hipStream_t stream) {
    (void)in_sizes; (void)n_in; (void)out_size; (void)ws_size;

    const float* v    = (const float*)d_in[0];
    const int*   ei   = (const int*)  d_in[1];   // int32 (JAX default demotes int64)
    const float* W1   = (const float*)d_in[2];
    const float* b1   = (const float*)d_in[3];
    const float* W2   = (const float*)d_in[4];
    const float* b2   = (const float*)d_in[5];
    const float* W3   = (const float*)d_in[6];
    const float* b3   = (const float*)d_in[7];
    const float* Wmu  = (const float*)d_in[8];
    const float* bmu  = (const float*)d_in[9];
    const float* Wstd = (const float*)d_in[10];
    const float* bstd = (const float*)d_in[11];

    const int* srcp = ei;
    const int* dstp = ei + NEDGES;

    char* ws = (char*)d_ws;
    size_t off = 0;
    auto take = [&](size_t bytes) -> char* {
        char* p = ws + off;
        off += (bytes + 255) & ~(size_t)255;
        return p;
    };
    float*          dis = (float*)take((size_t)NNODES * 4);
    float*          nrm = (float*)take((size_t)NEDGES * 4);
    unsigned short* xb  = (unsigned short*)take((size_t)NNODES * 512 * 2);
    float*          P   = (float*)take((size_t)NNODES * 512 * 4);   // GEMM output h
    float*          Q   = (float*)take((size_t)NNODES * 512 * 4);   // agg / activations
    unsigned short* wt  = (unsigned short*)take((size_t)442368 * 2);

    // bf16 transposed weight offsets (elements)
    unsigned short* wt1 = wt;
    unsigned short* wt2 = wt + 16384;
    unsigned short* wt3 = wt + 49152;
    unsigned short* wtm = wt + 180224;
    unsigned short* wts = wt + 311296;

    const int T = 256;
    auto blks = [](long n, int t) { return (unsigned)((n + t - 1) / t); };

    // degree / norm setup
    k_deg_init <<<blks(NNODES, T), T, 0, stream>>>(dis, NNODES);
    k_deg_accum<<<blks(NEDGES, T), T, 0, stream>>>(dstp, dis, NEDGES);
    k_rsqrt    <<<blks(NNODES, T), T, 0, stream>>>(dis, NNODES);
    k_norm     <<<blks(NEDGES, T), T, 0, stream>>>(srcp, dstp, dis, nrm, NEDGES);

    // weight convert + transpose
    k_wt<<<blks(128 * 128, T), T, 0, stream>>>(W1,   wt1, 128, 128);
    k_wt<<<blks(128 * 256, T), T, 0, stream>>>(W2,   wt2, 128, 256);
    k_wt<<<blks(256 * 512, T), T, 0, stream>>>(W3,   wt3, 256, 512);
    k_wt<<<blks(512 * 256, T), T, 0, stream>>>(Wmu,  wtm, 512, 256);
    k_wt<<<blks(512 * 256, T), T, 0, stream>>>(Wstd, wts, 512, 256);

    auto run_gcn = [&](const unsigned short* wtl, const float* bias,
                       int K, int Nout, float* outbuf, int relu) {
        unsigned waves = (unsigned)((NNODES / 16) * (Nout / 64));   // 16x64 strips
        k_gemm<<<(waves + 7) / 8, T, 0, stream>>>(xb, wtl, P, NNODES, K, Nout);
        long tot = (long)NNODES * Nout;
        k_selfloop<<<blks(tot, T), T, 0, stream>>>(P, dis, Q, NNODES, Nout);
        k_scatter <<<(NEDGES + 7) / 8, T, 0, stream>>>(srcp, dstp, nrm, P, Q, NEDGES, Nout);
        k_bias    <<<blks(tot, T), T, 0, stream>>>(Q, bias, outbuf, tot, Nout, relu);
    };

    // layer 1: v -> Q
    k_cvt<<<blks((long)NNODES * 128, T), T, 0, stream>>>(v, xb, (long)NNODES * 128);
    run_gcn(wt1, b1, 128, 128, Q, 1);
    // layer 2: Q -> Q
    k_cvt<<<blks((long)NNODES * 128, T), T, 0, stream>>>(Q, xb, (long)NNODES * 128);
    run_gcn(wt2, b2, 128, 256, Q, 1);
    // layer 3: Q -> Q
    k_cvt<<<blks((long)NNODES * 256, T), T, 0, stream>>>(Q, xb, (long)NNODES * 256);
    run_gcn(wt3, b3, 256, 512, Q, 1);

    // heads share x3 (already in Q) -> convert once
    k_cvt<<<blks((long)NNODES * 512, T), T, 0, stream>>>(Q, xb, (long)NNODES * 512);
    float* out = (float*)d_out;
    run_gcn(wtm, bmu,  512, 256, out, 0);                          // mu
    run_gcn(wts, bstd, 512, 256, out + (size_t)NNODES * 256, 0);   // std
}
